// Graph_based_MLP_85856396247985
// MI455X (gfx1250) — matrix operand order repaired and checked by reading, not compiled, not run on hardware
//
#include <hip/hip_runtime.h>
#include <hip/hip_bf16.h>
#include <math.h>

// ---------------------------------------------------------------------------
// CDNA5 (gfx1250) fused GCN-MLP. wave32, WMMA 16x16x32 bf16.
// All WMMA operands are stored in LDS in the major order their fragment
// layout wants, so every fragment load is 2x ds_load_b128 per lane.
// ---------------------------------------------------------------------------

typedef __attribute__((ext_vector_type(16))) __bf16 v16bf;
typedef __attribute__((ext_vector_type(8)))  float  v8f;

union FragBF { v16bf v; unsigned short u[16]; uint4 q[2]; };

static constexpr int NN   = 166;   // real node count
static constexpr int NPM  = 176;   // padded node rows of A-hat (11 * 16)
static constexpr int NPK  = 192;   // padded node K dim (6 * 32)
static constexpr int HID  = 64;
static constexpr int EMB  = 128;
static constexpr int LOWD = 64;
static constexpr int TILE_B = 2;
static constexpr float BN_EPS = 1e-5f;

__device__ __forceinline__ unsigned short f2bf(float f) {
  unsigned int u = __float_as_uint(f);
  u += 0x7FFFu + ((u >> 16) & 1u);          // round-to-nearest-even
  return (unsigned short)(u >> 16);
}
__device__ __forceinline__ unsigned int pack2bf(float lo, float hi) {
  return (unsigned int)f2bf(lo) | ((unsigned int)f2bf(hi) << 16);
}
__device__ __forceinline__ float relu_f(float v) { return v > 0.f ? v : 0.f; }

// A-operand: 16x32 bf16 tile at (row0,k0) of row-major [M][K] src.
// Per lane: u[0..7] = row[k0+8h .. +7], u[8..15] = row[k0+8h+16 .. +23].
// Both runs are 16B-aligned for our strides -> 2x ds_load_b128.
__device__ __forceinline__ v16bf load_a_bf16(const unsigned short* src, int row0,
                                             int k0, int stride, int lane) {
  FragBF f;
  const int m = lane & 15, half = lane >> 4;
  const unsigned short* rp = src + (row0 + m) * stride + k0 + half * 8;
  f.q[0] = ((const uint4*)rp)[0];   // elements 0..7
  f.q[1] = ((const uint4*)rp)[2];   // elements 16..23
  return f.v;
}

// B-operand: 32x16 bf16 tile at (k0,n0) of a TRANSPOSED [N][K] src (strideK).
// Per lane n: u[0..7] = K[k0+8h .. +7], u[8..15] = K[k0+8h+16 .. +23] -> 2x ds_load_b128.
__device__ __forceinline__ v16bf load_b_bf16_tr(const unsigned short* srcT, int k0,
                                                int n0, int strideK, int lane) {
  FragBF f;
  const int n = lane & 15, half = lane >> 4;
  const unsigned short* p = srcT + (n0 + n) * strideK + k0 + 8 * half;
  f.q[0] = ((const uint4*)p)[0];
  f.q[1] = ((const uint4*)p)[2];
  return f.v;
}

__device__ __forceinline__ v8f wmma_bf16(v16bf a, v16bf b, v8f c) {
  return __builtin_amdgcn_wmma_f32_16x16x32_bf16(false, a, false, b,
                                                 (short)0, c, false, false);
}

// ---------------------------------------------------------------------------
// K1: dense normalized adjacency (self loops) + bf16 weight prep.
// W_low / W_conv2 are stored TRANSPOSED ([out][in]) for B-operand loads.
// ---------------------------------------------------------------------------
__global__ __launch_bounds__(256) void prep_kernel(
    const int* __restrict__ edges, int E,
    const float* __restrict__ W_low, const float* __restrict__ W_conv2,
    float* __restrict__ A_f32, float* __restrict__ deg,
    unsigned short* __restrict__ A_bf, unsigned short* __restrict__ WlowT_bf,
    unsigned short* __restrict__ W2T_bf) {
  const int tid = threadIdx.x;
  for (int i = tid; i < NPM * NPK; i += 256) A_f32[i] = 0.f;
  for (int i = tid; i < NPM; i += 256) deg[i] = 0.f;
  __syncthreads();
  for (int e = tid; e < E; e += 256) atomicAdd(&deg[edges[E + e]], 1.0f);
  for (int n = tid; n < NN; n += 256) atomicAdd(&deg[n], 1.0f);  // self loop
  __syncthreads();
  for (int e = tid; e < E; e += 256) {
    const int s = edges[e], d = edges[E + e];
    atomicAdd(&A_f32[d * NPK + s], rsqrtf(deg[s]) * rsqrtf(deg[d]));
  }
  for (int n = tid; n < NN; n += 256) atomicAdd(&A_f32[n * NPK + n], 1.0f / deg[n]);
  __syncthreads();
  for (int i = tid; i < NPM * NPK; i += 256) A_bf[i] = f2bf(A_f32[i]);
  for (int i = tid; i < LOWD * EMB; i += 256) {          // WlowT[j][i] = W_low[i][j]
    const int j = i >> 6, c = i & 63;                    // j<128 out, c<64 in
    WlowT_bf[j * LOWD + c] = f2bf(W_low[c * EMB + j]);
  }
  for (int i = tid; i < HID * HID; i += 256) {           // W2T[j][i] = W2[i][j]
    const int j = i >> 6, c = i & 63;
    W2T_bf[j * HID + c] = f2bf(W_conv2[c * HID + j]);
  }
}

// ---------------------------------------------------------------------------
// K2/K4: per-column batch statistics (biased variance, matching jnp.var).
// ---------------------------------------------------------------------------
__global__ __launch_bounds__(256) void colstats_kernel(
    const float* __restrict__ x, int rows, int cols,
    float* __restrict__ mu, float* __restrict__ var) {
  const int c = blockIdx.x, tid = threadIdx.x;
  float s1 = 0.f, s2 = 0.f;
  for (int r = tid; r < rows; r += 256) {
    const float v = x[(size_t)r * cols + c];
    s1 += v; s2 += v * v;
  }
  __shared__ float sh1[256], sh2[256];
  sh1[tid] = s1; sh2[tid] = s2; __syncthreads();
  for (int off = 128; off > 0; off >>= 1) {
    if (tid < off) { sh1[tid] += sh1[tid + off]; sh2[tid] += sh2[tid + off]; }
    __syncthreads();
  }
  if (tid == 0) {
    const float m = sh1[0] / rows;
    mu[c] = m;
    var[c] = sh2[0] / rows - m * m;
  }
}

// ---------------------------------------------------------------------------
// K3: P = relu( BN(low) @ W_low + bl_low )  [B,128], WMMA bf16 GEMM.
// ---------------------------------------------------------------------------
__global__ __launch_bounds__(256) void lowmlp_kernel(
    const float* __restrict__ low, const float* __restrict__ mu1,
    const float* __restrict__ var1, const float* __restrict__ g_bn,
    const float* __restrict__ b_bn, const unsigned short* __restrict__ WlowT_bf,
    const float* __restrict__ bl_low, float* __restrict__ P) {
  __shared__ unsigned short Alds[128 * LOWD];     // [row][k] row-major (A operand)
  __shared__ unsigned short Wlds[EMB * LOWD];     // [n][k] transposed (B operand)
  const int tid = threadIdx.x, lane = tid & 31, wave = tid >> 5;
  const int row0 = blockIdx.x * 128;

  for (int i = tid; i < 128 * LOWD; i += 256) {
    const int r = i >> 6, c = i & 63;
    float v = low[(size_t)(row0 + r) * LOWD + c];
    v = (v - mu1[c]) * rsqrtf(var1[c] + BN_EPS) * g_bn[c] + b_bn[c];
    Alds[i] = f2bf(v);
  }
  {
    const unsigned int* s = (const unsigned int*)WlowT_bf;
    unsigned int* d = (unsigned int*)Wlds;
    for (int i = tid; i < (EMB * LOWD) / 2; i += 256) d[i] = s[i];
  }
  __syncthreads();

  const int mrow = wave * 16;
  for (int nt = 0; nt < 8; ++nt) {
    v8f acc = {0.f, 0.f, 0.f, 0.f, 0.f, 0.f, 0.f, 0.f};
#pragma unroll
    for (int ks = 0; ks < 2; ++ks) {
      const v16bf a = load_a_bf16(Alds, mrow, ks * 32, LOWD, lane);
      const v16bf b = load_b_bf16_tr(Wlds, ks * 32, nt * 16, LOWD, lane);
      acc = wmma_bf16(a, b, acc);
    }
    const int col = nt * 16 + (lane & 15);
    const int rb  = mrow + 8 * (lane >> 4);
    const float bias = bl_low[col];
#pragma unroll
    for (int r = 0; r < 8; ++r)
      P[(size_t)(row0 + rb + r) * EMB + col] = relu_f(acc[r] + bias);
  }
}

// ---------------------------------------------------------------------------
// K5: fused GCN (dense A-hat WMMA x2) + mean pool + head + log_softmax.
// LDS buffers per sample: buf0 = H0^T[feat][node] then T^T[feat][node];
//                         buf1 = H1[node][feat].
// ---------------------------------------------------------------------------
__global__ __launch_bounds__(256) void gcn_fused_kernel(
    const float* __restrict__ high, const unsigned short* __restrict__ A_bf,
    const unsigned short* __restrict__ W2T_bf, const float* __restrict__ W1,
    const float* __restrict__ b1, const float* __restrict__ b2,
    const float* __restrict__ P, const float* __restrict__ mu2,
    const float* __restrict__ var2, const float* __restrict__ g_low,
    const float* __restrict__ b_low, const float* __restrict__ W_fc1,
    const float* __restrict__ b_fc1, const float* __restrict__ W_cls,
    const float* __restrict__ b_cls, float* __restrict__ out) {
  extern __shared__ unsigned char smem_raw[];
  unsigned short* A_l  = (unsigned short*)smem_raw;              // [176][192] row-major
  unsigned short* W2_l = A_l + NPM * NPK;                        // [64][64] transposed
  unsigned short* Hbuf = W2_l + HID * HID;                       // TB * 2 * 192*64
  float* gpool = (float*)(Hbuf + TILE_B * 2 * NPK * HID);        // TB*64
  float* comb  = gpool + TILE_B * HID;                           // TB*192
  float* hmid  = comb + TILE_B * (HID + EMB);                    // TB*128
  float* lgts  = hmid + TILE_B * EMB;                            // TB*2
  float* b1l   = lgts + TILE_B * 2;                              // 64
  float* b2l   = b1l + HID;                                      // 64

  const int tid = threadIdx.x, lane = tid & 31, wave = tid >> 5;
  const int bbase = blockIdx.x * TILE_B;
  const int HS = NPK * HID;  // per-sample buffer elems

  // --- stage A-hat, W2^T, biases; compute H0^T = (x @ W1)^T (K=3, VALU) -----
  {
    const unsigned int* As = (const unsigned int*)A_bf;
    unsigned int* Ad = (unsigned int*)A_l;
    for (int i = tid; i < (NPM * NPK) / 2; i += 256) Ad[i] = As[i];
    const unsigned int* Ws = (const unsigned int*)W2T_bf;
    unsigned int* Wd = (unsigned int*)W2_l;
    for (int i = tid; i < (HID * HID) / 2; i += 256) Wd[i] = Ws[i];
  }
  if (tid < HID) { b1l[tid] = b1[tid]; b2l[tid] = b2[tid]; }
  for (int idx = tid; idx < TILE_B * HS; idx += 256) {
    const int s = idx / HS, rem = idx % HS;
    const int f = rem / NPK, n = rem % NPK;      // H0^T[feat][node]
    float v = 0.f;
    if (n < NN) {
      const float* xr = high + (size_t)(bbase + s) * 498 + n * 3;
      v = xr[0] * W1[0 * HID + f] + xr[1] * W1[1 * HID + f] + xr[2] * W1[2 * HID + f];
    }
    Hbuf[s * 2 * HS + rem] = f2bf(v);            // buffer 0 of sample s
  }
  if (tid < TILE_B * HID) gpool[tid] = 0.f;
  __syncthreads();

  // --- conv1 aggregate, transposed formulation: H1^T = H0^T x A-hat^T -------
  // A operand = H0^T row-major; B operand = A-hat row-major read as [n][k].
  // D (rows=feat, cols=node) is packed per lane into row-major H1[node][feat]
  // with a single b128 store.
  for (int task = wave; task < TILE_B * 4 * 11; task += 8) {
    const int s = task / 44, rem = task % 44, ft = rem / 11, nt = rem % 11;
    const unsigned short* Hs = Hbuf + (s * 2 + 0) * HS;   // H0^T
    v8f acc = {0.f, 0.f, 0.f, 0.f, 0.f, 0.f, 0.f, 0.f};
#pragma unroll
    for (int ks = 0; ks < 6; ++ks) {
      const v16bf a = load_a_bf16(Hs, ft * 16, ks * 32, NPK, lane);
      const v16bf b = load_b_bf16_tr(A_l, ks * 32, nt * 16, NPK, lane);
      acc = wmma_bf16(a, b, acc);
    }
    const int node = nt * 16 + (lane & 15);
    const int fb   = ft * 16 + 8 * (lane >> 4);
    unsigned short* Hd = Hbuf + (s * 2 + 1) * HS;         // H1[node][feat]
    uint4 q;
    q.x = pack2bf(relu_f(acc[0] + b1l[fb + 0]), relu_f(acc[1] + b1l[fb + 1]));
    q.y = pack2bf(relu_f(acc[2] + b1l[fb + 2]), relu_f(acc[3] + b1l[fb + 3]));
    q.z = pack2bf(relu_f(acc[4] + b1l[fb + 4]), relu_f(acc[5] + b1l[fb + 5]));
    q.w = pack2bf(relu_f(acc[6] + b1l[fb + 6]), relu_f(acc[7] + b1l[fb + 7]));
    *(uint4*)(Hd + node * HID + fb) = q;
  }
  __syncthreads();

  // --- T = H1 @ W2 (standard orientation), stored transposed T^T[feat][node]
  for (int task = wave; task < TILE_B * 11 * 4; task += 8) {
    const int s = task / 44, rem = task % 44, mt = rem / 4, nt = rem % 4;
    const unsigned short* Hs = Hbuf + (s * 2 + 1) * HS;   // H1 row-major (A)
    v8f acc = {0.f, 0.f, 0.f, 0.f, 0.f, 0.f, 0.f, 0.f};
#pragma unroll
    for (int ks = 0; ks < 2; ++ks) {
      const v16bf a = load_a_bf16(Hs, mt * 16, ks * 32, HID, lane);
      const v16bf b = load_b_bf16_tr(W2_l, ks * 32, nt * 16, HID, lane);
      acc = wmma_bf16(a, b, acc);
    }
    const int fo = nt * 16 + (lane & 15);                 // featOut (col)
    const int nb = mt * 16 + 8 * (lane >> 4);             // node row base
    unsigned short* Td = Hbuf + (s * 2 + 0) * HS;         // T^T[feat][node]
    uint4 q;
    q.x = pack2bf(acc[0], acc[1]);
    q.y = pack2bf(acc[2], acc[3]);
    q.z = pack2bf(acc[4], acc[5]);
    q.w = pack2bf(acc[6], acc[7]);
    *(uint4*)(Td + fo * NPK + nb) = q;
  }
  __syncthreads();

  // --- conv2 aggregate + relu + mean pool (LDS float atomics) ---------------
  for (int task = wave; task < TILE_B * 11 * 4; task += 8) {
    const int s = task / 44, rem = task % 44, mt = rem / 4, nt = rem % 4;
    const unsigned short* Ts = Hbuf + (s * 2 + 0) * HS;   // T^T (B operand)
    v8f acc = {0.f, 0.f, 0.f, 0.f, 0.f, 0.f, 0.f, 0.f};
#pragma unroll
    for (int ks = 0; ks < 6; ++ks) {
      const v16bf a = load_a_bf16(A_l, mt * 16, ks * 32, NPK, lane);
      const v16bf b = load_b_bf16_tr(Ts, ks * 32, nt * 16, NPK, lane);
      acc = wmma_bf16(a, b, acc);
    }
    const int col = nt * 16 + (lane & 15);                // feat
    const int rb  = mt * 16 + 8 * (lane >> 4);            // node row base
    const float bias = b2l[col];
#pragma unroll
    for (int r = 0; r < 8; ++r) {
      if (rb + r < NN) atomicAdd(&gpool[s * HID + col], relu_f(acc[r] + bias));
    }
  }
  __syncthreads();

  // --- head: combined = [pool/166, BN2(P)]; fc1 + relu; cls; log_softmax ----
  for (int idx = tid; idx < TILE_B * (HID + EMB); idx += 256) {
    const int s = idx / (HID + EMB), i = idx % (HID + EMB);
    float v;
    if (i < HID) {
      v = gpool[s * HID + i] * (1.0f / (float)NN);
    } else {
      const int j = i - HID;
      const float p = P[(size_t)(bbase + s) * EMB + j];
      v = (p - mu2[j]) * rsqrtf(var2[j] + BN_EPS) * g_low[j] + b_low[j];
    }
    comb[idx] = v;
  }
  __syncthreads();
  for (int idx = tid; idx < TILE_B * EMB; idx += 256) {
    const int s = idx / EMB, j = idx % EMB;
    float acc = b_fc1[j];
    for (int i = 0; i < HID + EMB; ++i)
      acc += comb[s * (HID + EMB) + i] * W_fc1[i * EMB + j];
    hmid[idx] = relu_f(acc);
  }
  __syncthreads();
  if (tid < TILE_B * 2) {
    const int s = tid >> 1, o = tid & 1;
    float acc = b_cls[o];
    for (int j = 0; j < EMB; ++j) acc += hmid[s * EMB + j] * W_cls[j * 2 + o];
    lgts[tid] = acc;
  }
  __syncthreads();
  if (tid < TILE_B * 2) {
    const int s = tid >> 1, o = tid & 1;
    const float l0 = lgts[s * 2 + 0], l1 = lgts[s * 2 + 1];
    const float m = fmaxf(l0, l1);
    const float lse = m + logf(expf(l0 - m) + expf(l1 - m));
    out[(size_t)(bbase + s) * 2 + o] = lgts[s * 2 + o] - lse;
  }
}

// ---------------------------------------------------------------------------
extern "C" void kernel_launch(void* const* d_in, const int* in_sizes, int n_in,
                              void* d_out, int out_size, void* d_ws, size_t ws_size,
                              hipStream_t stream) {
  const float* high      = (const float*)d_in[0];
  const float* low       = (const float*)d_in[1];
  const int*   edges     = (const int*)d_in[2];
  const float* g_bn_low  = (const float*)d_in[3];
  const float* b_bn_low  = (const float*)d_in[4];
  const float* W_low     = (const float*)d_in[5];
  const float* bl_low    = (const float*)d_in[6];
  const float* g_low     = (const float*)d_in[7];
  const float* b_low     = (const float*)d_in[8];
  const float* W_conv1   = (const float*)d_in[9];
  const float* b_conv1   = (const float*)d_in[10];
  const float* W_conv2   = (const float*)d_in[11];
  const float* b_conv2   = (const float*)d_in[12];
  const float* W_fc1     = (const float*)d_in[13];
  const float* b_fc1     = (const float*)d_in[14];
  const float* W_cls     = (const float*)d_in[15];
  const float* b_cls     = (const float*)d_in[16];

  const int B = in_sizes[1] / LOWD;  // 8192
  const int E = in_sizes[2] / 2;     // 4000

  // workspace layout (float offsets)
  float* ws = (float*)d_ws;
  float* A_f32 = ws;                                  // 33792
  float* deg   = ws + 33792;                          // 176
  float* mu1   = ws + 33968;                          // 64
  float* var1  = ws + 34032;                          // 64
  float* mu2   = ws + 34096;                          // 128
  float* var2  = ws + 34224;                          // 128
  unsigned short* A_bf     = (unsigned short*)(ws + 34368);  // 33792 u16
  unsigned short* WlowT_bf = (unsigned short*)(ws + 51264);  // 8192  u16
  unsigned short* W2T_bf   = (unsigned short*)(ws + 55360);  // 4096  u16
  float* P = ws + 57408;                              // B * 128

  prep_kernel<<<1, 256, 0, stream>>>(edges, E, W_low, W_conv2,
                                     A_f32, deg, A_bf, WlowT_bf, W2T_bf);
  colstats_kernel<<<LOWD, 256, 0, stream>>>(low, B, LOWD, mu1, var1);
  lowmlp_kernel<<<B / 128, 256, 0, stream>>>(low, mu1, var1, g_bn_low, b_bn_low,
                                             WlowT_bf, bl_low, P);
  colstats_kernel<<<EMB, 256, 0, stream>>>(P, B, EMB, mu2, var2);

  const size_t smem =
      (size_t)(NPM * NPK + HID * HID + TILE_B * 2 * NPK * HID) * sizeof(unsigned short) +
      (size_t)(TILE_B * HID + TILE_B * (HID + EMB) + TILE_B * EMB + TILE_B * 2 +
               2 * HID) * sizeof(float);
  (void)hipFuncSetAttribute((const void*)gcn_fused_kernel,
                            hipFuncAttributeMaxDynamicSharedMemorySize, (int)smem);
  gcn_fused_kernel<<<B / TILE_B, 256, smem, stream>>>(
      high, A_bf, W2T_bf, W_conv1, b_conv1, b_conv2, P, mu2, var2, g_low, b_low,
      W_fc1, b_fc1, W_cls, b_cls, (float*)d_out);
}
